// OnmiNERModel_18751827214626
// MI455X (gfx1250) — compile-verified, matrix operand order checked by reading, not run on hardware
//
#include <hip/hip_runtime.h>
#include <math.h>

// ---------------------------------------------------------------------------
// ArcFace span-classification loss for MI455X (gfx1250, wave32, WMMA).
// Pipeline: span-mean+LN+L2norm -> weight L2norm -> f16 WMMA GEMM (cosine)
//           -> fused ArcFace margin + log-softmax + mean NLL.
// GEMM is register-blocked 32x64 per wave (8 WMMA tiles) for A/B fragment
// reuse: 12 b128 loads per 8 v_wmma ops instead of 4 loads per op.
// ---------------------------------------------------------------------------

typedef __attribute__((ext_vector_type(16))) _Float16 v16h;
typedef __attribute__((ext_vector_type(8)))  _Float16 v8h;
typedef __attribute__((ext_vector_type(4)))  _Float16 v4h;
typedef __attribute__((ext_vector_type(8)))  float    v8f;

#define H_DIM     1024
#define N_CLS     4096
#define NEG_INF   (-3.402823466e38f)

// ArcFace constants (margin = 0.5, scale = 30)
#define ARC_SCALE 30.0f
#define ARC_COS_M 0.87758256189037271612f   // cos(0.5)
#define ARC_SIN_M 0.47942553860420300027f   // sin(0.5)
#define ARC_TH   (-0.87758256189037271612f) // cos(pi - 0.5)
#define ARC_MM    0.23971276930210150013f   // sin(pi - 0.5) * 0.5
#define LN_EPS    1e-7f

// ---------------- reductions (wave32) ----------------
__device__ __forceinline__ float waveReduceSum(float v) {
  #pragma unroll
  for (int off = 16; off > 0; off >>= 1) v += __shfl_xor(v, off, 32);
  return v;
}
__device__ __forceinline__ float waveReduceMax(float v) {
  #pragma unroll
  for (int off = 16; off > 0; off >>= 1) v = fmaxf(v, __shfl_xor(v, off, 32));
  return v;
}
__device__ __forceinline__ float blockReduceSum(float v, float* lds) {
  const int lane = threadIdx.x & 31, wid = threadIdx.x >> 5;
  v = waveReduceSum(v);
  if (lane == 0) lds[wid] = v;
  __syncthreads();
  const int nw = blockDim.x >> 5;
  float r = (threadIdx.x < (unsigned)nw) ? lds[threadIdx.x] : 0.0f;
  if (wid == 0) r = waveReduceSum(r);
  if (threadIdx.x == 0) lds[0] = r;
  __syncthreads();
  r = lds[0];
  __syncthreads();
  return r;
}
__device__ __forceinline__ float blockReduceMax(float v, float* lds) {
  const int lane = threadIdx.x & 31, wid = threadIdx.x >> 5;
  v = waveReduceMax(v);
  if (lane == 0) lds[wid] = v;
  __syncthreads();
  const int nw = blockDim.x >> 5;
  float r = (threadIdx.x < (unsigned)nw) ? lds[threadIdx.x] : NEG_INF;
  if (wid == 0) r = waveReduceMax(r);
  if (threadIdx.x == 0) lds[0] = r;
  __syncthreads();
  r = lds[0];
  __syncthreads();
  return r;
}

// ---------------- kernel 1: span mean -> LayerNorm -> L2 norm -> f16 ----------------
// grid = B*NS blocks, 256 threads; each thread owns 4 contiguous H columns.
__global__ void span_ln_l2_kernel(const float* __restrict__ enc,   // [B,S,H]
                                  const float* __restrict__ gamma, // [H]
                                  const float* __restrict__ beta,  // [H]
                                  const int*   __restrict__ heads, // [B*NS]
                                  const int*   __restrict__ tails, // [B*NS]
                                  _Float16*    __restrict__ embn,  // [B*NS, H] f16, L2-normalized
                                  int S, int NS) {
  __shared__ float lds[8];
  const int span = blockIdx.x;
  const int b    = span / NS;
  const int head = heads[span];
  const int tail = tails[span];
  const int col  = threadIdx.x << 2;

  const float* base = enc + (size_t)b * S * H_DIM + col;
  float a0 = 0.f, a1 = 0.f, a2 = 0.f, a3 = 0.f;
  for (int t = head; t < tail; ++t) {
    const float4 v = *(const float4*)(base + (size_t)t * H_DIM);
    a0 += v.x; a1 += v.y; a2 += v.z; a3 += v.w;
  }
  const float inv_cnt = 1.0f / (float)(tail - head);
  a0 *= inv_cnt; a1 *= inv_cnt; a2 *= inv_cnt; a3 *= inv_cnt;

  const float mean = blockReduceSum(a0 + a1 + a2 + a3, lds) * (1.0f / H_DIM);
  const float d0 = a0 - mean, d1 = a1 - mean, d2 = a2 - mean, d3 = a3 - mean;
  const float var = blockReduceSum(d0*d0 + d1*d1 + d2*d2 + d3*d3, lds) * (1.0f / H_DIM);
  const float rstd = rsqrtf(var + LN_EPS);

  const float4 g  = *(const float4*)(gamma + col);
  const float4 be = *(const float4*)(beta  + col);
  const float y0 = d0 * rstd * g.x + be.x;
  const float y1 = d1 * rstd * g.y + be.y;
  const float y2 = d2 * rstd * g.z + be.z;
  const float y3 = d3 * rstd * g.w + be.w;

  const float ss  = blockReduceSum(y0*y0 + y1*y1 + y2*y2 + y3*y3, lds);
  const float inv = 1.0f / fmaxf(sqrtf(ss), 1e-12f);

  v4h o;
  o[0] = (_Float16)(y0 * inv); o[1] = (_Float16)(y1 * inv);
  o[2] = (_Float16)(y2 * inv); o[3] = (_Float16)(y3 * inv);
  *(v4h*)(embn + (size_t)span * H_DIM + col) = o;
}

// ---------------- kernel 2: arc_weight row L2 norm -> f16 ----------------
// grid = N_CLS blocks, 256 threads, 4 columns/thread.
__global__ void wnorm_kernel(const float* __restrict__ w,   // [N_CLS, H]
                             _Float16*    __restrict__ wn)  // [N_CLS, H] f16
{
  __shared__ float lds[8];
  const int row = blockIdx.x;
  const int col = threadIdx.x << 2;
  const float4 v = *(const float4*)(w + (size_t)row * H_DIM + col);
  const float ss  = blockReduceSum(v.x*v.x + v.y*v.y + v.z*v.z + v.w*v.w, lds);
  const float inv = 1.0f / fmaxf(sqrtf(ss), 1e-12f);
  v4h o;
  o[0] = (_Float16)(v.x * inv); o[1] = (_Float16)(v.y * inv);
  o[2] = (_Float16)(v.z * inv); o[3] = (_Float16)(v.w * inv);
  *(v4h*)(wn + (size_t)row * H_DIM + col) = o;
}

// ---------------- kernel 3: cosine GEMM via v_wmma_f32_16x16x32_f16 ----------------
// C[M,N] = A[M,K] * Bm[N,K]^T.  One wave computes a 32x64 block (2x4 WMMA tiles)
// so each A fragment feeds 4 WMMAs and each B fragment feeds 2.
// A-operand layout (16-bit 16x32): lane<16 -> row M=lane, K in {0..7,16..23};
// lane>=16 -> row M=lane-16, K in {8..15,24..31}.  B symmetric with N.
__device__ __forceinline__ v16h load_frag(const _Float16* __restrict__ p) {
  const v8h lo = *(const v8h*)p;          // K+0..7 (or +8..15 for hi lanes)
  const v8h hi = *(const v8h*)(p + 16);   // K+16..23 (or +24..31)
  return __builtin_shufflevector(lo, hi, 0,1,2,3,4,5,6,7,8,9,10,11,12,13,14,15);
}

__global__ void gemm_wmma_kernel(const _Float16* __restrict__ A,   // [M,K] f16 row-major
                                 const _Float16* __restrict__ Bm,  // [N,K] f16 row-major (acts as B^T)
                                 float* __restrict__ C,            // [M,N] f32
                                 int M, int N, int K) {
  const int lane   = threadIdx.x & 31;
  const int wave   = threadIdx.x >> 5;
  const int wPerB  = blockDim.x >> 5;
  const int tile   = blockIdx.x * wPerB + wave;
  const int tilesN = N >> 6;              // 64-wide N blocks
  const int tm = (tile / tilesN) << 5;    // 32-row M block
  const int tn = (tile % tilesN) << 6;    // 64-col N block
  if (tm >= M) return;  // grid covers exactly; keeps EXEC all-1s in live waves

  const int r  = lane & 15;   // row within tile (M for A, N for B)
  const int hi = lane >> 4;   // K-half select

  const _Float16* aPtr0 = A + (size_t)(tm + r) * K + hi * 8;
  const _Float16* aPtr1 = aPtr0 + (size_t)16 * K;
  const _Float16* bPtr0 = Bm + (size_t)(tn + r) * K + hi * 8;
  const _Float16* bPtr1 = bPtr0 + (size_t)16 * K;
  const _Float16* bPtr2 = bPtr0 + (size_t)32 * K;
  const _Float16* bPtr3 = bPtr0 + (size_t)48 * K;

  v8f acc[2][4];
  #pragma unroll
  for (int i = 0; i < 2; ++i)
    #pragma unroll
    for (int j = 0; j < 4; ++j) acc[i][j] = (v8f){};

  for (int k = 0; k < K; k += 32) {
    const v16h a0 = load_frag(aPtr0 + k);
    const v16h a1 = load_frag(aPtr1 + k);
    const v16h b0 = load_frag(bPtr0 + k);
    const v16h b1 = load_frag(bPtr1 + k);
    const v16h b2 = load_frag(bPtr2 + k);
    const v16h b3 = load_frag(bPtr3 + k);

    acc[0][0] = __builtin_amdgcn_wmma_f32_16x16x32_f16(false, a0, false, b0, (short)0, acc[0][0], false, false);
    acc[0][1] = __builtin_amdgcn_wmma_f32_16x16x32_f16(false, a0, false, b1, (short)0, acc[0][1], false, false);
    acc[0][2] = __builtin_amdgcn_wmma_f32_16x16x32_f16(false, a0, false, b2, (short)0, acc[0][2], false, false);
    acc[0][3] = __builtin_amdgcn_wmma_f32_16x16x32_f16(false, a0, false, b3, (short)0, acc[0][3], false, false);
    acc[1][0] = __builtin_amdgcn_wmma_f32_16x16x32_f16(false, a1, false, b0, (short)0, acc[1][0], false, false);
    acc[1][1] = __builtin_amdgcn_wmma_f32_16x16x32_f16(false, a1, false, b1, (short)0, acc[1][1], false, false);
    acc[1][2] = __builtin_amdgcn_wmma_f32_16x16x32_f16(false, a1, false, b2, (short)0, acc[1][2], false, false);
    acc[1][3] = __builtin_amdgcn_wmma_f32_16x16x32_f16(false, a1, false, b3, (short)0, acc[1][3], false, false);
  }

  // C/D layout: lane<16 -> N=lane, VGPR i -> M=i; lane>=16 -> M=i+8.
  #pragma unroll
  for (int i = 0; i < 2; ++i) {
    #pragma unroll
    for (int j = 0; j < 4; ++j) {
      float* cPtr = C + (size_t)(tm + i * 16 + hi * 8) * N + tn + j * 16 + r;
      #pragma unroll
      for (int v = 0; v < 8; ++v) cPtr[(size_t)v * N] = acc[i][j][v];
    }
  }
}

// ---------------- kernel 4: ArcFace margin + log-softmax + mean NLL ----------------
// grid = M blocks (one per embedding row). Margin touches only the label column,
// so fold it into the row logsumexp instead of materializing logits.
__global__ void arcface_nll_kernel(const float* __restrict__ cosine, // [M, N_CLS]
                                   const int*   __restrict__ labels, // [M]
                                   float* __restrict__ out,          // [1]
                                   int M) {
  __shared__ float lds[8];
  const int row = blockIdx.x;
  const int lab = labels[row];
  const float* c = cosine + (size_t)row * N_CLS;

  const float cl  = c[lab];
  const float sl  = sqrtf(fmaxf(1.0f - cl * cl, 0.0f));
  const float phi = cl * ARC_COS_M - sl * ARC_SIN_M;
  const float adj = (cl > ARC_TH) ? phi : (cl - ARC_MM);

  float vmax = NEG_INF;
  for (int j = threadIdx.x; j < N_CLS; j += blockDim.x) {
    const float v = (j == lab) ? adj : c[j];
    vmax = fmaxf(vmax, v);
  }
  vmax = blockReduceMax(vmax, lds);

  float s = 0.0f;
  for (int j = threadIdx.x; j < N_CLS; j += blockDim.x) {
    const float v = (j == lab) ? adj : c[j];
    s += expf((v - vmax) * ARC_SCALE);
  }
  s = blockReduceSum(s, lds);

  if (threadIdx.x == 0) {
    // nll = logsumexp(logits) - logit[label] ; logits = SCALE * v
    const float nll = (vmax * ARC_SCALE + logf(s)) - adj * ARC_SCALE;
    atomicAdd(out, nll * (1.0f / (float)M));
  }
}

// ---------------------------------------------------------------------------
extern "C" void kernel_launch(void* const* d_in, const int* in_sizes, int n_in,
                              void* d_out, int out_size, void* d_ws, size_t ws_size,
                              hipStream_t stream) {
  const float* enc    = (const float*)d_in[0];  // [B,S,H]
  const float* gamma  = (const float*)d_in[1];  // [H]
  const float* beta   = (const float*)d_in[2];  // [H]
  const float* arcw   = (const float*)d_in[3];  // [N_CLS,H]
  const int*   heads  = (const int*)d_in[4];    // [B,NS]
  const int*   tails  = (const int*)d_in[5];    // [B,NS]
  const int*   labels = (const int*)d_in[6];    // [B,NS]

  const int B  = 16, S = 2048, NS = 64;
  const int M  = B * NS;  // 1024 span embeddings

  // workspace: emb_n f16 (2MB) | w_n f16 (8MB) | cosine f32 (16MB)
  _Float16* embn   = (_Float16*)d_ws;
  _Float16* wn     = embn + (size_t)M * H_DIM;
  float*    cosine = (float*)(wn + (size_t)N_CLS * H_DIM);

  hipMemsetAsync(d_out, 0, sizeof(float), stream);

  span_ln_l2_kernel<<<M, 256, 0, stream>>>(enc, gamma, beta, heads, tails, embn, S, NS);
  wnorm_kernel<<<N_CLS, 256, 0, stream>>>(arcw, wn);

  const int tiles = (M >> 5) * (N_CLS >> 6);        // 32 * 64 = 2048 waves
  gemm_wmma_kernel<<<tiles / 8, 256, 0, stream>>>(embn, wn, cosine, M, N_CLS, H_DIM);

  arcface_nll_kernel<<<M, 256, 0, stream>>>(cosine, labels, (float*)d_out, M);
}
